// NeRFSmall_45835890983142
// MI455X (gfx1250) — matrix operand order repaired and checked by reading, not compiled, not run on hardware
//
#include <hip/hip_runtime.h>
#include <hip/hip_bf16.h>
#include <math.h>

// ---------------------------------------------------------------------------
// Fully-fused NeRF-small MLP on CDNA5 (gfx1250), wave32 WMMA f16->f32.
// Orientation: per layer compute  H_out^T = W^T x H_in^T
//   A operand = W^T tile (16 out-features x 32 in-features), register resident
//   B operand = activation^T (32 in-features x 16 points)
//   D         = 16 out-features x 16 points (f32)
// D->B chaining: f16 pack + packed-f16 relu + v_permlanex16 half-swap.
// Input loads are software-pipelined one tile ahead of the WMMA chain.
// ---------------------------------------------------------------------------

typedef __attribute__((ext_vector_type(16))) _Float16 v16h;
typedef __attribute__((ext_vector_type(2)))  _Float16 v2h;
typedef __attribute__((ext_vector_type(8)))  float    v8f;
typedef __attribute__((ext_vector_type(8)))  unsigned v8u;

__device__ __forceinline__ v8f wmma16(v16h a, v16h b, v8f c) {
  // D = A(16x32 f16) * B(32x16 f16) + C(16x16 f32)
  return __builtin_amdgcn_wmma_f32_16x16x32_f16(false, a, false, b,
                                                (short)0, c, false, false);
}

__device__ __forceinline__ unsigned pk2(float a, float b) {
  v2h h; h[0] = (_Float16)a; h[1] = (_Float16)b;
  return __builtin_bit_cast(unsigned, h);
}

// Packed f16 ReLU on a cvt_pk result: single v_pk_max_num_f16 (operand is
// known-canonical, so no canonicalize companion op is emitted).
__device__ __forceinline__ unsigned pk2_relu(float a, float b) {
  v2h h; h[0] = (_Float16)a; h[1] = (_Float16)b;
  v2h z = {};
  v2h r = __builtin_elementwise_max(h, z);
  return __builtin_bit_cast(unsigned, r);
}

// lane L <-> lane L^16 exchange, pure VALU (v_permlanex16_b32 identity sel).
__device__ __forceinline__ unsigned swap16(unsigned v) {
  return __builtin_amdgcn_permlanex16(v, v, 0x76543210u, 0xFEDCBA98u,
                                      false, false);
}

// Load one A fragment of W^T (row-major W[K_in][N_out] in global memory).
// A fragment (this code's contiguous k order): lane p=L%16 is row M (= out feature f*16+p);
// laneLo halves i -> k = i (0..15); laneHi -> +16. Chunk c adds 32c.
__device__ __forceinline__ v16h load_wfrag(const float* __restrict__ W,
                                           int K_in, int N_out,
                                           int f, int c, int p, int hi) {
  v16h r;
#pragma unroll
  for (int i = 0; i < 16; ++i) {
    int k = i + hi * 16 + c * 32;
    int n = f * 16 + p;
    float v = 0.f;
    if (k < K_in && n < N_out) v = W[k * N_out + n];
    r[i] = (_Float16)v;
  }
  return r;
}

// Build B fragment for K-chunk c from D tiles 2c (Dl) and 2c+1 (Dh),
// applying ReLU in packed f16 before the half-swap.
// B fragment, same contiguous k order as A: lanes 0-15 hold k=0..15, lanes 16-31 hold k=16..31.
// laneLo: dwords0-3 = own Dl, dwords4-7 = partner Dl
// laneHi: dwords0-3 = partner Dh, dwords4-7 = own Dh
__device__ __forceinline__ v16h make_chunkB(v8f Dl, v8f Dh, bool lo) {
  unsigned ol[4], oh[4];
#pragma unroll
  for (int q = 0; q < 4; ++q) {
    ol[q] = pk2_relu(Dl[2*q], Dl[2*q+1]);
    oh[q] = pk2_relu(Dh[2*q], Dh[2*q+1]);
  }
  v8u bu;
#pragma unroll
  for (int q = 0; q < 4; ++q) {
    unsigned pl = swap16(ol[q]);
    unsigned ph = swap16(oh[q]);
    bu[q]     = lo ? ol[q] : ph;
    bu[4 + q] = lo ? pl    : oh[q];
  }
  return __builtin_bit_cast(v16h, bu);
}

__global__ __launch_bounds__(256)
void nerf_fused_kernel(const float* __restrict__ x,
                       const float* __restrict__ ws0,
                       const float* __restrict__ ws1,
                       const float* __restrict__ ws2,
                       const float* __restrict__ wc0,
                       const float* __restrict__ wc1,
                       const float* __restrict__ wc2,
                       const float* __restrict__ wc3,
                       float* __restrict__ out, int n) {
  const int lane = (int)(threadIdx.x & 31u);
  const int p    = lane & 15;
  const int hi   = lane >> 4;
  const bool lo  = (hi == 0);

  // -------- register-resident weight fragments (W^T as A operands) --------
  v16h As0[4], As1[4][2], As2[2], Ac0[4], Ac1[4][2], Ac2[4][2], Ac3[2];
#pragma unroll
  for (int f = 0; f < 4; ++f) {
    As0[f] = load_wfrag(ws0,  3, 64, f, 0, p, hi);   // K padded 3 -> 32
    Ac0[f] = load_wfrag(wc0, 18, 64, f, 0, p, hi);   // K padded 18 -> 32
#pragma unroll
    for (int c = 0; c < 2; ++c) {
      As1[f][c] = load_wfrag(ws1, 64, 64, f, c, p, hi);
      Ac1[f][c] = load_wfrag(wc1, 64, 64, f, c, p, hi);
      Ac2[f][c] = load_wfrag(wc2, 64, 64, f, c, p, hi);
    }
  }
#pragma unroll
  for (int c = 0; c < 2; ++c) {
    As2[c] = load_wfrag(ws2, 64, 16, 0, c, p, hi);   // 64 -> 16 (sigma+geo)
    Ac3[c] = load_wfrag(wc3, 64,  3, 0, c, p, hi);   // 64 -> 3 (cols padded)
  }

  const int wave   = (int)((blockIdx.x * blockDim.x + threadIdx.x) >> 5);
  const int nwaves = (int)((gridDim.x * blockDim.x) >> 5);
  const int ntiles = n >> 4;
  const v8f zero = {};

  // Software-pipelined input: loaded 1 iteration ahead of use.
  float2 xa, xb, xc;
  if (wave < ntiles) {
    const float2* xp =
        reinterpret_cast<const float2*>(x + ((size_t)wave * 16 + p) * 6);
    xa = xp[0]; xb = xp[1]; xc = xp[2];
  }

  for (int t = wave; t < ntiles; t += nwaves) {
    const float x0 = xa.x, x1 = xa.y, x2 = xb.x;
    const float x3 = xb.y, x4 = xc.x, x5 = xc.y;

    // Prefetch next tile's inputs; the 36-WMMA chain below hides the latency.
    const int tn = t + nwaves;
    if (tn < ntiles) {
      const float2* xp =
          reinterpret_cast<const float2*>(x + ((size_t)tn * 16 + p) * 6);
      xa = xp[0]; xb = xp[1]; xc = xp[2];
    }

    // ---------------- sigma net: 3 -> 64 -> 64 -> 16 ----------------
    // B = X^T, K padded 3 -> 32: only laneLo dwords 0-1 nonzero.
    v8u bsu = {};
    bsu[0] = lo ? pk2(x0, x1) : 0u;
    bsu[1] = lo ? pk2(x2, 0.f) : 0u;
    v16h Bs = __builtin_bit_cast(v16h, bsu);

    v8f D[4];
#pragma unroll
    for (int f = 0; f < 4; ++f) D[f] = wmma16(As0[f], Bs, zero);

    v16h B0 = make_chunkB(D[0], D[1], lo);
    v16h B1 = make_chunkB(D[2], D[3], lo);

#pragma unroll
    for (int f = 0; f < 4; ++f)
      D[f] = wmma16(As1[f][1], B1, wmma16(As1[f][0], B0, zero));

    B0 = make_chunkB(D[0], D[1], lo);
    B1 = make_chunkB(D[2], D[3], lo);

    v8f Ds2 = wmma16(As2[1], B1, wmma16(As2[0], B0, zero));
    float sigma_pre = Ds2[0];           // laneLo j=0: pre-softplus sigma

    // ---------- color net input: concat(views(3), geo_feat(15)) ----------
    // geo_feat = s2 outputs 1..15 (NO activation). K padded 18 -> 32.
    // laneLo k: 0-2 views, 3-9 own outputs 1-7, 10-15 partner outputs 8-13.
    // laneHi k: 16-17 own outputs 14-15, rest zero.
    unsigned g01 = pk2(Ds2[0], Ds2[1]);   // outputs {0,1} (this half)
    unsigned g23 = pk2(Ds2[2], Ds2[3]);
    unsigned g45 = pk2(Ds2[4], Ds2[5]);
    unsigned g67 = pk2(Ds2[6], Ds2[7]);
    unsigned p01 = swap16(g01);           // partner outputs {8,9} for laneLo
    unsigned p23 = swap16(g23);
    unsigned p45 = swap16(g45);

    v8u bcu;
    bcu[0] = lo ? pk2(x3, x4)         : g67;              // k0,1 | k16,17
    bcu[1] = lo ? pk2(x5, Ds2[1])     : 0u;               // k2,3
    bcu[2] = lo ? pk2(Ds2[2], Ds2[3]) : 0u;               // k4,5
    bcu[3] = lo ? pk2(Ds2[4], Ds2[5]) : 0u;               // k6,7
    bcu[4] = lo ? g67 : 0u;                               // k8,9
    bcu[5] = lo ? p01 : 0u;                               // k10,11
    bcu[6] = lo ? p23 : 0u;                               // k12,13
    bcu[7] = lo ? p45 : 0u;                               // k14,15
    v16h Bc = __builtin_bit_cast(v16h, bcu);

    // ---------------- color net: 18 -> 64 -> 64 -> 64 -> 3 ----------------
#pragma unroll
    for (int f = 0; f < 4; ++f) D[f] = wmma16(Ac0[f], Bc, zero);

    B0 = make_chunkB(D[0], D[1], lo);
    B1 = make_chunkB(D[2], D[3], lo);
#pragma unroll
    for (int f = 0; f < 4; ++f)
      D[f] = wmma16(Ac1[f][1], B1, wmma16(Ac1[f][0], B0, zero));

    B0 = make_chunkB(D[0], D[1], lo);
    B1 = make_chunkB(D[2], D[3], lo);
#pragma unroll
    for (int f = 0; f < 4; ++f)
      D[f] = wmma16(Ac2[f][1], B1, wmma16(Ac2[f][0], B0, zero));

    B0 = make_chunkB(D[0], D[1], lo);
    B1 = make_chunkB(D[2], D[3], lo);
    v8f Dc3 = wmma16(Ac3[1], B1, wmma16(Ac3[0], B0, zero));

    // ------------- epilogue: [color0..2, softplus(sigma)] -------------
    float sp = (sigma_pre > 20.f) ? sigma_pre : log1pf(expf(sigma_pre));
    if (lo) {
      float4 o = make_float4(Dc3[0], Dc3[1], Dc3[2], sp);
      *reinterpret_cast<float4*>(out + ((size_t)t * 16 + p) * 4) = o;
    }
  }
}

extern "C" void kernel_launch(void* const* d_in, const int* in_sizes, int n_in,
                              void* d_out, int out_size, void* d_ws, size_t ws_size,
                              hipStream_t stream) {
  const float* x   = (const float*)d_in[0];
  const float* ws0 = (const float*)d_in[1];
  const float* ws1 = (const float*)d_in[2];
  const float* ws2 = (const float*)d_in[3];
  const float* wc0 = (const float*)d_in[4];
  const float* wc1 = (const float*)d_in[5];
  const float* wc2 = (const float*)d_in[6];
  const float* wc3 = (const float*)d_in[7];
  float* out = (float*)d_out;

  const int n = in_sizes[0] / 6;           // number of points (1M)
  const int threads = 256;                 // 8 waves per workgroup
  const int blocks  = 1024;                // 8192 waves, grid-stride over tiles

  nerf_fused_kernel<<<blocks, threads, 0, stream>>>(
      x, ws0, ws1, ws2, wc0, wc1, wc2, wc3, out, n);
}
